// MultiHeadAttention_encoder_85469849190481
// MI455X (gfx1250) — matrix operand order repaired and checked by reading, not compiled, not run on hardware
//
#include <hip/hip_runtime.h>
#include <hip/hip_bf16.h>

#define BB 16
#define SS 1024
#define DD 256
#define HH 8
#define NH2 4
#define DH 128
#define DEPTH 32

typedef __attribute__((ext_vector_type(16))) __bf16 v16bf;
typedef __attribute__((ext_vector_type(8)))  __bf16 v8bf;
typedef __attribute__((ext_vector_type(8)))  float  v8f;
typedef __attribute__((ext_vector_type(4)))  float  f32x4;

static __device__ __forceinline__ v8f wmma_bf16(v16bf a, v16bf b, v8f c) {
  return __builtin_amdgcn_wmma_f32_16x16x32_bf16(false, a, false, b, (short)0, c, false, false);
}

static __device__ __forceinline__ v16bf cat8(v8bf lo, v8bf hi) {
  v16bf r;
#pragma unroll
  for (int e = 0; e < 8; ++e) { r[e] = lo[e]; r[e + 8] = hi[e]; }
  return r;
}

// put 8 f32 (two float4) into fragment positions [base..base+7] as bf16
static __device__ __forceinline__ void put8(v16bf& d, int base, f32x4 a, f32x4 b) {
#pragma unroll
  for (int e = 0; e < 4; ++e) {
    d[base + e]     = (__bf16)a[e];
    d[base + 4 + e] = (__bf16)b[e];
  }
}

// ---------------------------------------------------------------------------
// Weight transpose + bf16 convert:  WT[n*K + k] = W[k*N + n]
// ---------------------------------------------------------------------------
__global__ __launch_bounds__(256)
void transpose_w(const float* __restrict__ W, __bf16* __restrict__ WT, int K, int N) {
  int idx = blockIdx.x * 256 + threadIdx.x;
  if (idx >= K * N) return;
  int n = idx / K, k = idx - n * K;
  WT[idx] = (__bf16)W[(size_t)k * N + n];
}

// ---------------------------------------------------------------------------
// Projection GEMM: Y = X(f32, Mx256) @ W(256xN) + bias, via WT (bf16, Nx256).
// One wave -> 16x16 tile, 8x v_wmma_f32_16x16x32_bf16, all-b128 operand loads.
// storeT: store transposed [b][col][s] layout (used for V so PV B-fragments
//         become contiguous 32B loads).
// ---------------------------------------------------------------------------
__global__ __launch_bounds__(32)
void proj_gemm(const float* __restrict__ X, const __bf16* __restrict__ WT,
               const float* __restrict__ bias, __bf16* __restrict__ Y,
               int N, int storeT) {
  const int lane = threadIdx.x;
  const int half = lane >> 4;
  const int lid  = lane & 15;
  const int mt = blockIdx.x, nt = blockIdx.y;
  v8f acc = {};
  for (int kt = 0; kt < DD / 32; ++kt) {
    const float* xa = X + (size_t)(mt * 16 + lid) * DD + kt * 32;
    v16bf a;
    put8(a, 0, *(const f32x4*)(xa + half * 16),     *(const f32x4*)(xa + half * 16 + 4));
    put8(a, 8, *(const f32x4*)(xa + half * 16 + 8), *(const f32x4*)(xa + half * 16 + 12));
    v16bf b = *(const v16bf*)(WT + (size_t)(nt * 16 + lid) * DD + kt * 32 + half * 16);
    acc = wmma_bf16(a, b, acc);
  }
  float bv = bias[nt * 16 + lid];
  if (!storeT) {
#pragma unroll
    for (int r = 0; r < 8; ++r) {
      int m = r + 8 * half;
      Y[(size_t)(mt * 16 + m) * N + nt * 16 + lid] = (__bf16)(acc[r] + bv);
    }
  } else {  // Y[b][col][s], col in [0,N)
#pragma unroll
    for (int r = 0; r < 8; ++r) {
      int row = mt * 16 + r + 8 * half;
      int b_  = row >> 10, s = row & (SS - 1);
      Y[((size_t)b_ * N + nt * 16 + lid) * SS + s] = (__bf16)(acc[r] + bv);
    }
  }
}

// ---------------------------------------------------------------------------
// Output GEMM: out(f32) = concat(bf16, 16384x256) @ Wo + bo, via WoT(bf16).
// ---------------------------------------------------------------------------
__global__ __launch_bounds__(32)
void out_gemm(const __bf16* __restrict__ Xb, const __bf16* __restrict__ WoT,
              const float* __restrict__ bias, float* __restrict__ Y) {
  const int lane = threadIdx.x;
  const int half = lane >> 4;
  const int lid  = lane & 15;
  const int mt = blockIdx.x, nt = blockIdx.y;
  v8f acc = {};
  for (int kt = 0; kt < DD / 32; ++kt) {
    const __bf16* xa = Xb + (size_t)(mt * 16 + lid) * DD + kt * 32;
    v16bf a = cat8(*(const v8bf*)(xa + half * 16), *(const v8bf*)(xa + half * 16 + 8));
    v16bf b = *(const v16bf*)(WoT + (size_t)(nt * 16 + lid) * DD + kt * 32 + half * 16);
    acc = wmma_bf16(a, b, acc);
  }
  float bv = bias[nt * 16 + lid];
#pragma unroll
  for (int r = 0; r < 8; ++r) {
    int m = r + 8 * half;
    Y[(size_t)(mt * 16 + m) * DD + nt * 16 + lid] = acc[r] + bv;
  }
}

// ---------------------------------------------------------------------------
// Fused attention: one wave per (b, branch, head, 16-row q tile).
//  P1: 64 QK wmma -> LDS logits (scaled + mask), K B-fragment = 1 v16bf load.
//  P2: per-row max/sum via b128 LDS scans + wave shuffles (+ dist stats).
//  P3: w = softmax * (adj | dist-softmax); float4 attn_w stream + LDS.
//  P4: 64 PV wmma; V^T layout makes B-fragments contiguous v16bf loads.
// ---------------------------------------------------------------------------
__global__ __launch_bounds__(32)
void attn_kernel(const __bf16* __restrict__ Qd, const __bf16* __restrict__ Kd,
                 const __bf16* __restrict__ VdT, const __bf16* __restrict__ Qa,
                 const __bf16* __restrict__ Ka, const __bf16* __restrict__ VaT,
                 const float* __restrict__ mask, const float* __restrict__ adj,
                 const float* __restrict__ dist, float* __restrict__ attnw,
                 __bf16* __restrict__ concat) {
  __shared__ float lds[16 * SS];  // 64 KB of the 320 KB WGP LDS
  const int lane = threadIdx.x;
  const int half = lane >> 4;
  const int lid  = lane & 15;
  const int bid  = blockIdx.x;
  const int qt     = bid & 63;
  const int h      = (bid >> 6) & 3;
  const int branch = (bid >> 8) & 1;  // 0 = dist, 1 = adj
  const int b      = bid >> 9;
  const int qrow0  = qt * 16;
  const float scale = 0.17677669529663687f;  // 1/sqrt(32)

  const __bf16* Q  = branch ? Qa : Qd;
  const __bf16* K  = branch ? Ka : Kd;
  const __bf16* VT = branch ? VaT : VdT;
  const __bf16* qbase = Q + ((size_t)(b * SS + qrow0 + lid)) * DH + h * DEPTH;
  const __bf16* vbase = VT + ((size_t)b * DH + h * DEPTH + lid) * SS;

  // A operand: Q tile 16x32 (depth == WMMA K) -> lane half holds depth[half*16 .. +15]
  v16bf aq = cat8(*(const v8bf*)(qbase + half * 16),
                  *(const v8bf*)(qbase + half * 16 + 8));

  // ---- Phase 1: logits -> LDS ----
  for (int kt = 0; kt < SS / 16; ++kt) {
    const __bf16* kbase = K + ((size_t)(b * SS + kt * 16 + lid)) * DH + h * DEPTH;
    __builtin_prefetch(kbase + 16 * DH, 0, 0);
    // B[kd][n]: lane n=lid holds K-row segment depth[half*16 .. +15] -> 32B
    v16bf bk = *(const v16bf*)(kbase + half * 16);
    v8f lg = {};
    lg = wmma_bf16(aq, bk, lg);
    int col = kt * 16 + lid;
    float mv = mask[b * SS + col] * -1e9f;
#pragma unroll
    for (int r = 0; r < 8; ++r)
      lds[(r + 8 * half) * SS + col] = lg[r] * scale + mv;
  }
  __syncthreads();

  // ---- Phase 2: row stats (lane r keeps stats for row r), b128 scans ----
  float st_max = 0.f, st_inv = 0.f, st_dmax = 0.f, st_dinv = 0.f;
  for (int r = 0; r < 16; ++r) {
    float mx = -1e30f;
    for (int c = lane * 4; c < SS; c += 128) {
      f32x4 v = *(const f32x4*)&lds[r * SS + c];
      mx = fmaxf(fmaxf(fmaxf(mx, v[0]), fmaxf(v[1], v[2])), v[3]);
    }
#pragma unroll
    for (int off = 16; off; off >>= 1) mx = fmaxf(mx, __shfl_xor(mx, off, 32));
    float sm = 0.f;
    for (int c = lane * 4; c < SS; c += 128) {
      f32x4 v = *(const f32x4*)&lds[r * SS + c];
      sm += __expf(v[0] - mx) + __expf(v[1] - mx) + __expf(v[2] - mx) + __expf(v[3] - mx);
    }
#pragma unroll
    for (int off = 16; off; off >>= 1) sm += __shfl_xor(sm, off, 32);

    float dmx = 0.f, dsm = 1.f;
    if (branch == 0) {  // head-independent softmax(dist + neg) stats
      const float* drow = dist + ((size_t)b * SS + qrow0 + r) * SS;
      dmx = -1e30f;
      for (int c = lane * 4; c < SS; c += 128) {
        f32x4 dv = *(const f32x4*)&drow[c];
        f32x4 mk = *(const f32x4*)&mask[b * SS + c];
#pragma unroll
        for (int e = 0; e < 4; ++e) dmx = fmaxf(dmx, dv[e] + mk[e] * -1e9f);
      }
#pragma unroll
      for (int off = 16; off; off >>= 1) dmx = fmaxf(dmx, __shfl_xor(dmx, off, 32));
      dsm = 0.f;
      for (int c = lane * 4; c < SS; c += 128) {
        f32x4 dv = *(const f32x4*)&drow[c];
        f32x4 mk = *(const f32x4*)&mask[b * SS + c];
#pragma unroll
        for (int e = 0; e < 4; ++e) dsm += __expf(dv[e] + mk[e] * -1e9f - dmx);
      }
#pragma unroll
      for (int off = 16; off; off >>= 1) dsm += __shfl_xor(dsm, off, 32);
    }
    if (lane == r) {
      st_max = mx; st_inv = 1.f / sm;
      st_dmax = dmx; st_dinv = 1.f / dsm;
    }
  }

  // ---- Phase 3: weights -> attn_w (b128 stream) + LDS ----
  const size_t attn_head = ((size_t)(b * HH + branch * NH2 + h)) * SS * SS;
  for (int r = 0; r < 16; ++r) {
    float mx   = __shfl(st_max, r, 32);
    float inv  = __shfl(st_inv, r, 32);
    float dmx  = __shfl(st_dmax, r, 32);
    float dinv = __shfl(st_dinv, r, 32);
    const size_t grow = (size_t)b * SS + qrow0 + r;
    float* orow = attnw + attn_head + (size_t)(qrow0 + r) * SS;
    for (int c = lane * 4; c < SS; c += 128) {
      f32x4 l4 = *(const f32x4*)&lds[r * SS + c];
      f32x4 w4;
      if (branch) {
        f32x4 a4 = *(const f32x4*)&adj[grow * SS + c];
#pragma unroll
        for (int e = 0; e < 4; ++e) w4[e] = __expf(l4[e] - mx) * inv * a4[e];
      } else {
        f32x4 d4 = *(const f32x4*)&dist[grow * SS + c];
        f32x4 m4 = *(const f32x4*)&mask[b * SS + c];
#pragma unroll
        for (int e = 0; e < 4; ++e)
          w4[e] = __expf(l4[e] - mx) * inv *
                  (__expf(d4[e] + m4[e] * -1e9f - dmx) * dinv);
      }
      *(f32x4*)&lds[r * SS + c] = w4;
      *(f32x4*)&orow[c] = w4;
    }
  }
  __syncthreads();

  // ---- Phase 4: O = W @ V (two accumulators: depth cols 0-15, 16-31) ----
  v8f acc0 = {}, acc1 = {};
  for (int ck = 0; ck < SS / 32; ++ck) {
    __builtin_prefetch(vbase + (ck + 1) * 32, 0, 0);
    // A = w chunk 16x32 from LDS (f32 -> bf16), two contiguous 8-float chunks
    const float* wrow = &lds[(size_t)lid * SS + ck * 32];
    v16bf aw;
    put8(aw, 0, *(const f32x4*)(wrow + half * 16),     *(const f32x4*)(wrow + half * 16 + 4));
    put8(aw, 8, *(const f32x4*)(wrow + half * 16 + 8), *(const f32x4*)(wrow + half * 16 + 12));
    // B from V^T: lane n=lid -> VT row segment keys[ck*32 + half*16 .. +15]
    v16bf b0 = *(const v16bf*)(vbase + ck * 32 + half * 16);
    v16bf b1 = *(const v16bf*)(vbase + 16 * SS + ck * 32 + half * 16);
    acc0 = wmma_bf16(aw, b0, acc0);
    acc1 = wmma_bf16(aw, b1, acc1);
  }
#pragma unroll
  for (int r = 0; r < 8; ++r) {
    int m = r + 8 * half;
    size_t off = ((size_t)(b * SS + qrow0 + m)) * DD + branch * DH + h * DEPTH;
    concat[off + lid]      = (__bf16)acc0[r];
    concat[off + 16 + lid] = (__bf16)acc1[r];
  }
}

// ---------------------------------------------------------------------------
extern "C" void kernel_launch(void* const* d_in, const int* in_sizes, int n_in,
                              void* d_out, int out_size, void* d_ws, size_t ws_size,
                              hipStream_t stream) {
  (void)in_sizes; (void)n_in; (void)out_size; (void)ws_size;
  const float* v_ori = (const float*)d_in[0];
  const float* k_ori = (const float*)d_in[1];
  const float* q_ori = (const float*)d_in[2];
  const float* mask  = (const float*)d_in[3];
  const float* adj   = (const float*)d_in[4];
  const float* dist  = (const float*)d_in[5];
  const float* Wq_d = (const float*)d_in[6];  const float* bq_d = (const float*)d_in[7];
  const float* Wk_d = (const float*)d_in[8];  const float* bk_d = (const float*)d_in[9];
  const float* Wv_d = (const float*)d_in[10]; const float* bv_d = (const float*)d_in[11];
  const float* Wq_a = (const float*)d_in[12]; const float* bq_a = (const float*)d_in[13];
  const float* Wk_a = (const float*)d_in[14]; const float* bk_a = (const float*)d_in[15];
  const float* Wv_a = (const float*)d_in[16]; const float* bv_a = (const float*)d_in[17];
  const float* Wo   = (const float*)d_in[18]; const float* bo   = (const float*)d_in[19];

  char* ws = (char*)d_ws;
  const size_t szQK = (size_t)BB * SS * DH * sizeof(__bf16);  // 4 MB each
  __bf16* qd  = (__bf16*)(ws + 0 * szQK);
  __bf16* kd  = (__bf16*)(ws + 1 * szQK);
  __bf16* vdT = (__bf16*)(ws + 2 * szQK);
  __bf16* qa  = (__bf16*)(ws + 3 * szQK);
  __bf16* ka  = (__bf16*)(ws + 4 * szQK);
  __bf16* vaT = (__bf16*)(ws + 5 * szQK);
  __bf16* concat = (__bf16*)(ws + 6 * szQK);          // 8 MB
  char*   wbase  = ws + 6 * szQK + (size_t)BB * SS * DD * sizeof(__bf16);
  const size_t szW = (size_t)DD * DH * sizeof(__bf16);  // 64 KB
  __bf16* WTq_d = (__bf16*)(wbase + 0 * szW);
  __bf16* WTk_d = (__bf16*)(wbase + 1 * szW);
  __bf16* WTv_d = (__bf16*)(wbase + 2 * szW);
  __bf16* WTq_a = (__bf16*)(wbase + 3 * szW);
  __bf16* WTk_a = (__bf16*)(wbase + 4 * szW);
  __bf16* WTv_a = (__bf16*)(wbase + 5 * szW);
  __bf16* WoT   = (__bf16*)(wbase + 6 * szW);           // 128 KB

  const int twg = (DD * DH + 255) / 256;
  transpose_w<<<twg, 256, 0, stream>>>(Wq_d, WTq_d, DD, DH);
  transpose_w<<<twg, 256, 0, stream>>>(Wk_d, WTk_d, DD, DH);
  transpose_w<<<twg, 256, 0, stream>>>(Wv_d, WTv_d, DD, DH);
  transpose_w<<<twg, 256, 0, stream>>>(Wq_a, WTq_a, DD, DH);
  transpose_w<<<twg, 256, 0, stream>>>(Wk_a, WTk_a, DD, DH);
  transpose_w<<<twg, 256, 0, stream>>>(Wv_a, WTv_a, DD, DH);
  transpose_w<<<(DD * DD + 255) / 256, 256, 0, stream>>>(Wo, WoT, DD, DD);

  dim3 pg(BB * SS / 16, DH / 16);
  proj_gemm<<<pg, 32, 0, stream>>>(q_ori, WTq_d, bq_d, qd,  DH, 0);
  proj_gemm<<<pg, 32, 0, stream>>>(k_ori, WTk_d, bk_d, kd,  DH, 0);
  proj_gemm<<<pg, 32, 0, stream>>>(v_ori, WTv_d, bv_d, vdT, DH, 1);
  proj_gemm<<<pg, 32, 0, stream>>>(q_ori, WTq_a, bq_a, qa,  DH, 0);
  proj_gemm<<<pg, 32, 0, stream>>>(k_ori, WTk_a, bk_a, ka,  DH, 0);
  proj_gemm<<<pg, 32, 0, stream>>>(v_ori, WTv_a, bv_a, vaT, DH, 1);

  float* out   = (float*)d_out;
  float* attnw = out + (size_t)BB * SS * DD;
  attn_kernel<<<BB * 2 * NH2 * (SS / 16), 32, 0, stream>>>(
      qd, kd, vdT, qa, ka, vaT, mask, adj, dist, attnw, concat);

  dim3 og(BB * SS / 16, DD / 16);
  out_gemm<<<og, 32, 0, stream>>>(concat, WoT, bo, out);
}
